// IN_ODEfunc_12034498363979
// MI455X (gfx1250) — matrix operand order repaired and checked by reading, not compiled, not run on hardware
//
#include <hip/hip_runtime.h>

typedef __attribute__((ext_vector_type(16))) _Float16 v16h;
typedef __attribute__((ext_vector_type(8)))  _Float16 v8h;
typedef __attribute__((ext_vector_type(8)))  float    v8f;

#define WPB 8               // waves per 256-thread block (node kernel)
#define EWPB 12             // waves per 384-thread block (edge kernel)
#define EDGE_ASTRIDE 168    // per-row halves in edge activation LDS tile (bank-friendly pad)
#define NODE_ASTRIDE 136    // per-row halves in node activation LDS tile

#define FR_PACKED_HALVES 92160   // fR: 5120 + 3*25600 + 10240
#define FO_PACKED_HALVES 9216    // fO: 7168 + 2048
#define TOT_PACKED_HALVES 101376
#define FR_BIAS_FLOATS 704       // 4*160 + 64
#define FO_BIAS_FLOATS 128       // 112 + 16
#define TOT_BIAS_FLOATS 832

static __device__ inline v8f wmma16(v16h a, v16h b, v8f c) {
  return __builtin_amdgcn_wmma_f32_16x16x32_f16(
      /*neg_a=*/false, a, /*neg_b=*/false, b,
      /*c_mod=*/(short)0, c, /*reuse_a=*/false, /*reuse_b=*/false);
}

// Read an A-fragment (16x32 f16) for one K-tile from an LDS activation tile.
// Lane l holds row = l&15; elems 0..7 -> K = kt*32 + kb + 0..7,
// elems 8..15 -> K = kt*32 + kb + 16..23, where kb = (l>=16) ? 8 : 0.
static __device__ inline v16h load_a_frag(const _Float16* act, int row,
                                          int stride, int kt, int hi) {
  int kb = kt * 32 + (hi ? 8 : 0);
  const _Float16* p = act + row * stride + kb;
  v8h lo = *(const v8h*)(p);
  v8h hh = *(const v8h*)(p + 16);
  v16h a;
#pragma unroll
  for (int i = 0; i < 8; ++i) { a[i] = lo[i]; a[i + 8] = hh[i]; }
  return a;
}

// D-tile epilogue: bias + ReLU + f16 convert + store column nt*16+row into act tile.
static __device__ inline void store_d_relu(_Float16* act, int rb, int col,
                                           float bias, v8f c) {
#pragma unroll
  for (int v = 0; v < 8; ++v)
    act[(rb + v) * EDGE_ASTRIDE + col] = (_Float16)fmaxf(c[v] + bias, 0.0f);
}

// Build layer-0 A fragment (13 inputs, K padded to 32) by gathering one edge.
static __device__ inline v16h gather_a0(const float* __restrict__ x,
                                        const float* __restrict__ ofx,
                                        const int* __restrict__ src,
                                        const int* __restrict__ tgt,
                                        int e, int M, float tt, int hi) {
  float f[13];
  if (e < M) {
    int s = src[e], g = tgt[e];
    float4 xs = ((const float4*)x)[s];
    float4 xg = ((const float4*)x)[g];
    float2 os = ((const float2*)ofx)[s];
    float2 og = ((const float2*)ofx)[g];
    f[0] = xs.x; f[1] = xs.y; f[2] = xs.z; f[3] = xs.w; f[4] = os.x; f[5] = os.y;
    f[6] = xg.x; f[7] = xg.y; f[8] = xg.z; f[9] = xg.w; f[10] = og.x; f[11] = og.y;
    f[12] = tt;
  } else {
#pragma unroll
    for (int i = 0; i < 13; ++i) f[i] = 0.0f;
  }
  v16h a = {};  // elems 8..15 map to K>=16 -> stay zero
  if (hi) {
    a[0] = (_Float16)f[8];  a[1] = (_Float16)f[9];  a[2] = (_Float16)f[10];
    a[3] = (_Float16)f[11]; a[4] = (_Float16)f[12];
  } else {
    a[0] = (_Float16)f[0]; a[1] = (_Float16)f[1]; a[2] = (_Float16)f[2];
    a[3] = (_Float16)f[3]; a[4] = (_Float16)f[4]; a[5] = (_Float16)f[5];
    a[6] = (_Float16)f[6]; a[7] = (_Float16)f[7];
  }
  return a;
}

// ---------------------------------------------------------------- prep ----
__global__ void zero_f32_kernel(float* __restrict__ p, int n) {
  int i = blockIdx.x * blockDim.x + threadIdx.x;
  if (i < n) p[i] = 0.0f;
}

// Repack f32 weights [di,do] row-major into f16 WMMA B-fragments.
// Fragment element: lane l, elem e -> K = kt*32 + 8*(l>=16) + (e<8 ? e : 8+e), N = nt*16 + (l&15).
__global__ void pack_w_kernel(const float* __restrict__ W0, const float* __restrict__ W1,
                              const float* __restrict__ W2, const float* __restrict__ W3,
                              const float* __restrict__ W4, const float* __restrict__ W5,
                              const float* __restrict__ W6, _Float16* __restrict__ wPacked) {
  int idx = blockIdx.x * blockDim.x + threadIdx.x;
  if (idx >= TOT_PACKED_HALVES) return;
  const int off[8] = {0, 5120, 30720, 56320, 81920, 92160, 99328, 101376};
  const int NT[7]  = {10, 10, 10, 10, 4, 7, 1};
  const int DI[7]  = {13, 150, 150, 150, 150, 50, 100};
  const int DO[7]  = {150, 150, 150, 150, 50, 100, 4};
  int L = 0;
  while (idx >= off[L + 1]) ++L;
  int rem = idx - off[L];
  int nt512 = NT[L] * 512;
  int kt = rem / nt512;  rem -= kt * nt512;
  int nt = rem / 512;    rem -= nt * 512;
  int lane = rem >> 4;
  int e = rem & 15;
  int K = kt * 32 + ((lane >= 16) ? 8 : 0) + ((e < 8) ? e : (e + 8));
  int n = nt * 16 + (lane & 15);
  const float* W = (L == 0) ? W0 : (L == 1) ? W1 : (L == 2) ? W2
                 : (L == 3) ? W3 : (L == 4) ? W4 : (L == 5) ? W5 : W6;
  float v = (K < DI[L] && n < DO[L]) ? W[K * DO[L] + n] : 0.0f;
  wPacked[idx] = (_Float16)v;
}

__global__ void pack_b_kernel(const float* __restrict__ B0, const float* __restrict__ B1,
                              const float* __restrict__ B2, const float* __restrict__ B3,
                              const float* __restrict__ B4, const float* __restrict__ B5,
                              const float* __restrict__ B6, float* __restrict__ bPacked) {
  int idx = blockIdx.x * blockDim.x + threadIdx.x;
  if (idx >= TOT_BIAS_FLOATS) return;
  const int off[8] = {0, 160, 320, 480, 640, 704, 816, 832};
  const int DO[7]  = {150, 150, 150, 150, 50, 100, 4};
  int L = 0;
  while (idx >= off[L + 1]) ++L;
  int j = idx - off[L];
  const float* B = (L == 0) ? B0 : (L == 1) ? B1 : (L == 2) ? B2
                 : (L == 3) ? B3 : (L == 4) ? B4 : (L == 5) ? B5 : B6;
  bPacked[idx] = (j < DO[L]) ? B[j] : 0.0f;
}

// ---------------------------------------------------------- edge kernel ----
// Per wave iteration: 32 edges = two 16-row M-tiles sharing every B fragment
// (each B LDS load feeds two independent WMMA accumulation chains c0/c1).
// fR MLP 13->150->150->150->150->50, f16 WMMA with f32 accumulate, weights
// LDS-resident, per-wave LDS activation tiles for the inter-layer transpose,
// atomic scatter-add of E into E'.
__global__ __launch_bounds__(384) void edge_mlp_scatter_kernel(
    const float* __restrict__ tptr, const float* __restrict__ x,
    const float* __restrict__ ofx, const int* __restrict__ src,
    const int* __restrict__ tgt, const _Float16* __restrict__ wPacked,
    const float* __restrict__ bPacked, float* __restrict__ ePrime, int M) {
  extern __shared__ uint4 smem_u4[];
  char* smem = (char*)smem_u4;
  _Float16* sW = (_Float16*)smem;                                  // 184320 B
  float* sB = (float*)(smem + FR_PACKED_HALVES * 2);               // 2816 B
  _Float16* sActAll = (_Float16*)(smem + FR_PACKED_HALVES * 2 + FR_BIAS_FLOATS * 4);

  {  // stage fR weights + biases into LDS once per block
    const uint4* g4 = (const uint4*)wPacked;
    uint4* s4 = (uint4*)sW;
    for (int i = threadIdx.x; i < FR_PACKED_HALVES / 8; i += blockDim.x) s4[i] = g4[i];
    for (int i = threadIdx.x; i < FR_BIAS_FLOATS; i += blockDim.x) sB[i] = bPacked[i];
  }
  __syncthreads();

  const int lane = threadIdx.x & 31;
  const int wave = threadIdx.x >> 5;
  const int hi   = lane >> 4;
  const int row  = lane & 15;
  const int rb   = hi ? 8 : 0;
  _Float16* act0 = sActAll + wave * (2 * 16 * EDGE_ASTRIDE);
  _Float16* act1 = act0 + 16 * EDGE_ASTRIDE;
  const float tt = tptr[0];
  const int numPairs = (M + 31) >> 5;
  const int wstride  = gridDim.x * EWPB;

  for (int pair = blockIdx.x * EWPB + wave; pair < numPairs; pair += wstride) {
    const int base = pair * 32;
    // ---- layer 0: gather inputs, KT=1, NT=10, bias+ReLU ----
    v16h A0a = gather_a0(x, ofx, src, tgt, base + row,      M, tt, hi);
    v16h A0b = gather_a0(x, ofx, src, tgt, base + 16 + row, M, tt, hi);
#pragma unroll
    for (int nt = 0; nt < 10; ++nt) {
      v16h b = *(const v16h*)(sW + (nt * 32 + lane) * 16);
      v8f c0 = {}, c1 = {};
      c0 = wmma16(A0a, b, c0);
      c1 = wmma16(A0b, b, c1);
      float bias = sB[nt * 16 + row];
      store_d_relu(act0, rb, nt * 16 + row, bias, c0);
      store_d_relu(act1, rb, nt * 16 + row, bias, c1);
    }

    // ---- layers 1..3: KT=5, NT=10, bias+ReLU ----
    int woff = 5120, boff = 160;
    for (int L = 1; L <= 3; ++L) {
      v16h Aa[5], Ab[5];
#pragma unroll
      for (int kt = 0; kt < 5; ++kt) {
        Aa[kt] = load_a_frag(act0, row, EDGE_ASTRIDE, kt, hi);
        Ab[kt] = load_a_frag(act1, row, EDGE_ASTRIDE, kt, hi);
      }
#pragma unroll
      for (int nt = 0; nt < 10; ++nt) {
        v8f c0 = {}, c1 = {};
#pragma unroll
        for (int kt = 0; kt < 5; ++kt) {
          v16h b = *(const v16h*)(sW + woff + ((kt * 10 + nt) * 32 + lane) * 16);
          c0 = wmma16(Aa[kt], b, c0);
          c1 = wmma16(Ab[kt], b, c1);
        }
        float bias = sB[boff + nt * 16 + row];
        store_d_relu(act0, rb, nt * 16 + row, bias, c0);
        store_d_relu(act1, rb, nt * 16 + row, bias, c1);
      }
      woff += 25600; boff += 160;
    }

    // ---- layer 4: KT=5, NT=4 (50 valid), linear, scatter-add to E' ----
    {
      v16h Aa[5], Ab[5];
#pragma unroll
      for (int kt = 0; kt < 5; ++kt) {
        Aa[kt] = load_a_frag(act0, row, EDGE_ASTRIDE, kt, hi);
        Ab[kt] = load_a_frag(act1, row, EDGE_ASTRIDE, kt, hi);
      }
      int tg0[8], tg1[8];
#pragma unroll
      for (int v = 0; v < 8; ++v) {
        int e0 = base + rb + v;
        int e1 = base + 16 + rb + v;
        tg0[v] = (e0 < M) ? tgt[e0] : -1;
        tg1[v] = (e1 < M) ? tgt[e1] : -1;
      }
#pragma unroll
      for (int nt = 0; nt < 4; ++nt) {
        v8f c0 = {}, c1 = {};
#pragma unroll
        for (int kt = 0; kt < 5; ++kt) {
          v16h b = *(const v16h*)(sW + 81920 + ((kt * 4 + nt) * 32 + lane) * 16);
          c0 = wmma16(Aa[kt], b, c0);
          c1 = wmma16(Ab[kt], b, c1);
        }
        int feat = nt * 16 + row;
        if (feat < 50) {
          float bias = sB[640 + feat];
#pragma unroll
          for (int v = 0; v < 8; ++v) {
            if (tg0[v] >= 0) atomicAdd(&ePrime[tg0[v] * 50 + feat], c0[v] + bias);
            if (tg1[v] >= 0) atomicAdd(&ePrime[tg1[v] * 50 + feat], c1[v] + bias);
          }
        }
      }
    }
  }
}

// ---------------------------------------------------------- node kernel ----
// fO MLP 50->100->4 per node (16 nodes per wave tile), same WMMA machinery.
__global__ __launch_bounds__(256) void node_mlp_kernel(
    const _Float16* __restrict__ wPacked, const float* __restrict__ bPacked,
    const float* __restrict__ ePrime, float* __restrict__ P, int N) {
  extern __shared__ uint4 smem_u4[];
  char* smem = (char*)smem_u4;
  _Float16* sW = (_Float16*)smem;                                  // 18432 B
  float* sB = (float*)(smem + FO_PACKED_HALVES * 2);               // 512 B
  _Float16* sActAll = (_Float16*)(smem + FO_PACKED_HALVES * 2 + FO_BIAS_FLOATS * 4);

  {
    const uint4* g4 = (const uint4*)(wPacked + FR_PACKED_HALVES);
    uint4* s4 = (uint4*)sW;
    for (int i = threadIdx.x; i < FO_PACKED_HALVES / 8; i += blockDim.x) s4[i] = g4[i];
    for (int i = threadIdx.x; i < FO_BIAS_FLOATS; i += blockDim.x)
      sB[i] = bPacked[FR_BIAS_FLOATS + i];
  }
  __syncthreads();

  const int lane = threadIdx.x & 31;
  const int wave = threadIdx.x >> 5;
  const int hi   = lane >> 4;
  const int row  = lane & 15;
  const int rb   = hi ? 8 : 0;
  _Float16* act  = sActAll + wave * 16 * NODE_ASTRIDE;
  // zero once: guarantees K in [112,128) reads as 0 for layer 1
  for (int i = lane; i < 16 * NODE_ASTRIDE; i += 32) act[i] = (_Float16)0.0f;

  const int numTiles = (N + 15) >> 4;
  const int wstride  = gridDim.x * WPB;
  for (int tile = blockIdx.x * WPB + wave; tile < numTiles; tile += wstride) {
    int n = tile * 16 + row;
    bool nvalid = (n < N);
    // ---- build A from E' (K: 50 padded to 64, KT=2) ----
    v16h A[2];
#pragma unroll
    for (int kt = 0; kt < 2; ++kt) {
      v16h a = {};
      int kb = kt * 32 + (hi ? 8 : 0);
#pragma unroll
      for (int i = 0; i < 8; ++i) {
        int K0 = kb + i, K1 = kb + 16 + i;
        a[i]     = (nvalid && K0 < 50) ? (_Float16)ePrime[n * 50 + K0] : (_Float16)0.0f;
        a[i + 8] = (nvalid && K1 < 50) ? (_Float16)ePrime[n * 50 + K1] : (_Float16)0.0f;
      }
      A[kt] = a;
    }
    // ---- layer 0: KT=2, NT=7, bias+ReLU ----
#pragma unroll
    for (int nt = 0; nt < 7; ++nt) {
      v8f c = {};
#pragma unroll
      for (int kt = 0; kt < 2; ++kt) {
        v16h b = *(const v16h*)(sW + ((kt * 7 + nt) * 32 + lane) * 16);
        c = wmma16(A[kt], b, c);
      }
      float bias = sB[nt * 16 + row];
#pragma unroll
      for (int v = 0; v < 8; ++v)
        act[(rb + v) * NODE_ASTRIDE + nt * 16 + row] = (_Float16)fmaxf(c[v] + bias, 0.0f);
    }
    // ---- layer 1: KT=4, NT=1, linear, write P ----
    {
      v8f c = {};
#pragma unroll
      for (int kt = 0; kt < 4; ++kt) {
        v16h a = load_a_frag(act, row, NODE_ASTRIDE, kt, hi);
        v16h b = *(const v16h*)(sW + 7168 + (kt * 32 + lane) * 16);
        c = wmma16(a, b, c);
      }
      if (row < 4) {
        float bias = sB[112 + row];
#pragma unroll
        for (int v = 0; v < 8; ++v) {
          int nn = tile * 16 + rb + v;
          if (nn < N) P[nn * 4 + row] = c[v] + bias;
        }
      }
    }
  }
}

// -------------------------------------------------------------- launch ----
extern "C" void kernel_launch(void* const* d_in, const int* in_sizes, int n_in,
                              void* d_out, int out_size, void* d_ws, size_t ws_size,
                              hipStream_t stream) {
  (void)n_in; (void)out_size; (void)ws_size;
  const float* t   = (const float*)d_in[0];
  const float* x   = (const float*)d_in[1];
  const float* ofx = (const float*)d_in[2];
  const int* src   = (const int*)d_in[3];
  const int* tgt   = (const int*)d_in[4];
  const float* W[7] = {(const float*)d_in[5],  (const float*)d_in[7],
                       (const float*)d_in[9],  (const float*)d_in[11],
                       (const float*)d_in[13], (const float*)d_in[15],
                       (const float*)d_in[17]};
  const float* B[7] = {(const float*)d_in[6],  (const float*)d_in[8],
                       (const float*)d_in[10], (const float*)d_in[12],
                       (const float*)d_in[14], (const float*)d_in[16],
                       (const float*)d_in[18]};
  float* P = (float*)d_out;

  const int N = in_sizes[1] / 4;   // 100000
  const int M = in_sizes[3];       // 2000000

  char* ws = (char*)d_ws;
  size_t epBytes = ((size_t)N * 50 * sizeof(float) + 255) & ~(size_t)255;
  float*     ePrime  = (float*)ws;
  _Float16*  wPacked = (_Float16*)(ws + epBytes);
  float*     bPacked = (float*)(ws + epBytes + (size_t)TOT_PACKED_HALVES * 2);

  const int nE = N * 50;
  zero_f32_kernel<<<(nE + 255) / 256, 256, 0, stream>>>(ePrime, nE);
  pack_w_kernel<<<(TOT_PACKED_HALVES + 255) / 256, 256, 0, stream>>>(
      W[0], W[1], W[2], W[3], W[4], W[5], W[6], wPacked);
  pack_b_kernel<<<(TOT_BIAS_FLOATS + 255) / 256, 256, 0, stream>>>(
      B[0], B[1], B[2], B[3], B[4], B[5], B[6], bPacked);

  const size_t edgeSmem = (size_t)FR_PACKED_HALVES * 2 + FR_BIAS_FLOATS * 4 +
                          (size_t)EWPB * 2 * 16 * EDGE_ASTRIDE * 2;  // 316160 B
  const size_t nodeSmem = (size_t)FO_PACKED_HALVES * 2 + FO_BIAS_FLOATS * 4 +
                          (size_t)WPB * 16 * NODE_ASTRIDE * 2;       // 53760 B

  int numPairs = (M + 31) >> 5;
  int edgeBlocks = (numPairs + EWPB - 1) / EWPB;
  if (edgeBlocks > 2048) edgeBlocks = 2048;
  edge_mlp_scatter_kernel<<<edgeBlocks, 384, edgeSmem, stream>>>(
      t, x, ofx, src, tgt, wPacked, bPacked, ePrime, M);

  int numNodeTiles = (N + 15) >> 4;
  int nodeBlocks = (numNodeTiles + WPB - 1) / WPB;
  if (nodeBlocks > 2048) nodeBlocks = 2048;
  node_mlp_kernel<<<nodeBlocks, 256, nodeSmem, stream>>>(
      wPacked, bPacked, ePrime, P, N);
}
